// FlagOS_ChunkGatedDelta_5523327943183
// MI455X (gfx1250) — matrix operand order repaired and checked by reading, not compiled, not run on hardware
//
#include <hip/hip_runtime.h>
#include <hip/hip_bf16.h>

typedef __attribute__((ext_vector_type(16))) __bf16 v16bf;
typedef __attribute__((ext_vector_type(8)))  __bf16 v8bf;
typedef __attribute__((ext_vector_type(4)))  __bf16 v4bf;
typedef __attribute__((ext_vector_type(8)))  float  v8f;

#define Bdim  4
#define Hdim  16
#define Tdim  4096
#define Ddim  64
#define CHUNK 64
#define LD    72   // fp32 LDS row stride
#define LDB   72   // bf16 LDS row stride (144B pitch -> conflict-free b128 lanes)

static __device__ __forceinline__ v16bf cat16(v8bf lo, v8bf hi) {
  return __builtin_shufflevector(lo, hi, 0, 1, 2, 3, 4, 5, 6, 7,
                                         8, 9, 10, 11, 12, 13, 14, 15);
}

// A fragment (16x32) from row-major bf16: two contiguous 8-elem runs per lane.
static __device__ __forceinline__ v16bf ldA(const __bf16* s, int mBase, int kBase, int lane) {
  const int m = mBase + (lane & 15), h = lane >> 4;
  const v8bf lo = *reinterpret_cast<const v8bf*>(s + m * LDB + kBase + h * 8);
  const v8bf hi = *reinterpret_cast<const v8bf*>(s + m * LDB + kBase + 16 + h * 8);
  return cat16(lo, hi);
}

// B fragment (32x16) where the SOURCE holds B transposed: st[n][k] = B[k][n].
// Two contiguous 8-elem runs per lane.
static __device__ __forceinline__ v16bf ldB(const __bf16* st, int kBase, int nBase, int lane) {
  const int n = nBase + (lane & 15), h = lane >> 4;
  const v8bf lo = *reinterpret_cast<const v8bf*>(st + n * LDB + kBase + h * 8);
  const v8bf hi = *reinterpret_cast<const v8bf*>(st + n * LDB + kBase + 16 + h * 8);
  return cat16(lo, hi);
}

static __device__ __forceinline__ v8f wmma_bf16(v16bf a, v16bf b, v8f c) {
  return __builtin_amdgcn_wmma_f32_16x16x32_bf16(false, a, false, b, (short)0, c, false, false);
}

// One workgroup per (b,h) head. Chunked delta rule with C=64.
__global__ __launch_bounds__(256) void delta_rule_chunked_kernel(
    const float* __restrict__ q, const float* __restrict__ k,
    const float* __restrict__ v, const float* __restrict__ beta,
    float* __restrict__ out)
{
  __shared__ float  sV[CHUNK * LD];    // V (async-copied) -> RHS -> U, fp32
  __shared__ float  sG[CHUNK * LD];    // G = K K^T, fp32 (substitution accuracy)
  __shared__ __bf16 sKb[CHUNK * LDB];  // K  row-major bf16 (A op; B op of K^T)
  __shared__ __bf16 sKt[CHUNK * LDB];  // K^T bf16: sKt[d][t] (A op for S update)
  __shared__ __bf16 sQb[CHUNK * LDB];  // Q  row-major bf16
  __shared__ __bf16 sSt[CHUNK * LDB];  // S^T bf16: sSt[e][d] (B op for @S)
  __shared__ __bf16 sUt[CHUNK * LDB];  // U^T bf16: sUt[e][t] (B op for @U)
  __shared__ __bf16 sPb[CHUNK * LDB];  // P = tril(Q K^T) row-major bf16
  __shared__ float  sBeta[CHUNK];

  const int bh   = blockIdx.x;   // 0..63 (= b*H + h)
  const int tid  = threadIdx.x;  // 0..255
  const int lane = tid & 31;
  const int wave = tid >> 5;     // 0..7

  const float* qh = q    + (size_t)bh * Tdim * Ddim;
  const float* kh = k    + (size_t)bh * Tdim * Ddim;
  const float* vh = v    + (size_t)bh * Tdim * Ddim;
  const float* bp = beta + (size_t)bh * Tdim;
  float*       oh = out  + (size_t)bh * Tdim * Ddim;

  for (int i = tid; i < CHUNK * LDB; i += 256) sSt[i] = (__bf16)0.f;
  __syncthreads();

  // Each wave owns two 16x16 tiles of the 4x4 tile grid; its S tiles live in
  // registers across the whole sequence (WMMA C operand).
  const int tt0 = wave * 2;
  v8f sacc[2] = {v8f{}, v8f{}};

  const unsigned long long vbase = (unsigned long long)(uintptr_t)vh;

  for (int ci = 0; ci < Tdim / CHUNK; ++ci) {
    const int t0 = ci * CHUNK;

    // ---- stage chunk: V async (raw copy), K/Q via VGPRs with one-time cvt ----
    for (int idx = tid; idx < CHUNK * Ddim / 4; idx += 256) {
      const int r = (idx * 4) / Ddim;
      const int c = (idx * 4) % Ddim;

      // async DMA of V bytes straight into LDS (no conversion needed)
      const unsigned ldsoff = (unsigned)(uintptr_t)&sV[r * LD + c];
      const unsigned goff   = (unsigned)(((t0 + r) * Ddim + c) * 4);
      asm volatile("global_load_async_to_lds_b128 %0, %1, %2"
                   :: "v"(ldsoff), "v"(goff), "s"(vbase) : "memory");

      const float4 fk = *reinterpret_cast<const float4*>(kh + (size_t)(t0 + r) * Ddim + c);
      const float4 fq = *reinterpret_cast<const float4*>(qh + (size_t)(t0 + r) * Ddim + c);
      v4bf kb; kb[0] = (__bf16)fk.x; kb[1] = (__bf16)fk.y; kb[2] = (__bf16)fk.z; kb[3] = (__bf16)fk.w;
      v4bf qb; qb[0] = (__bf16)fq.x; qb[1] = (__bf16)fq.y; qb[2] = (__bf16)fq.z; qb[3] = (__bf16)fq.w;
      *reinterpret_cast<v4bf*>(&sKb[r * LDB + c]) = kb;
      *reinterpret_cast<v4bf*>(&sQb[r * LDB + c]) = qb;
      sKt[(c + 0) * LDB + r] = kb[0];
      sKt[(c + 1) * LDB + r] = kb[1];
      sKt[(c + 2) * LDB + r] = kb[2];
      sKt[(c + 3) * LDB + r] = kb[3];
    }
    if (tid < CHUNK) sBeta[tid] = bp[t0 + tid];
    if (ci + 1 < Tdim / CHUNK) {  // prefetch next chunk while computing
      const size_t nb = (size_t)(t0 + CHUNK) * Ddim + (size_t)tid * 16;
      __builtin_prefetch(qh + nb, 0, 1);
      __builtin_prefetch(kh + nb, 0, 1);
      __builtin_prefetch(vh + nb, 0, 1);
    }
    asm volatile("s_wait_asynccnt 0" ::: "memory");
    __syncthreads();

    // ---- M = K @ S0 (B = sSt) and G = K @ K^T (B = sKb as K^T) ----
    //      RHS = beta*(V - M) in place in sV; G -> sG
#pragma unroll
    for (int u = 0; u < 2; ++u) {
      const int tt = tt0 + u;
      const int tm = (tt >> 2) * 16, tn = (tt & 3) * 16;
      v8f cm = {}; v8f cg = {};
#pragma unroll
      for (int k0 = 0; k0 < Ddim; k0 += 32) {
        const v16bf ak = ldA(sKb, tm, k0, lane);
        cm = wmma_bf16(ak, ldB(sSt, k0, tn, lane), cm);
        cg = wmma_bf16(ak, ldB(sKb, k0, tn, lane), cg);
      }
      const int n  = tn + (lane & 15);
      const int mb = tm + ((lane >> 4) << 3);
#pragma unroll
      for (int r = 0; r < 8; ++r) {
        const int t = mb + r;
        sV[t * LD + n] = sBeta[t] * (sV[t * LD + n] - cm[r]);
        sG[t * LD + n] = cg[r];
      }
    }
    __syncthreads();

    // ---- forward substitution in fp32: (I + tril(beta*G, strict)) U = RHS.
    // Thread e owns column e (no cross-lane deps); emits bf16 U^T as it goes.
    if (tid < CHUNK) {
      const int e = tid;
      sUt[e * LDB + 0] = (__bf16)sV[0 * LD + e];
      for (int t = 1; t < CHUNK; ++t) {
        float acc = sV[t * LD + e];
        const float bt = sBeta[t];
        for (int s = 0; s < t; ++s)
          acc -= bt * sG[t * LD + s] * sV[s * LD + e];
        sV[t * LD + e] = acc;
        sUt[e * LDB + t] = (__bf16)acc;
      }
    }
    __syncthreads();

    // ---- P = tril(Q @ K^T, incl diag) -> sPb (bf16 row-major) ----
#pragma unroll
    for (int u = 0; u < 2; ++u) {
      const int tt = tt0 + u;
      const int tm = (tt >> 2) * 16, tn = (tt & 3) * 16;
      v8f cp = {};
#pragma unroll
      for (int k0 = 0; k0 < Ddim; k0 += 32)
        cp = wmma_bf16(ldA(sQb, tm, k0, lane), ldB(sKb, k0, tn, lane), cp);
      const int n  = tn + (lane & 15);
      const int mb = tm + ((lane >> 4) << 3);
#pragma unroll
      for (int r = 0; r < 8; ++r) {
        const int t = mb + r;
        sPb[t * LDB + n] = (__bf16)((n <= t) ? cp[r] : 0.f);
      }
    }
    __syncthreads();

    // ---- O = Q @ S0 + P @ U -> global ; S += K^T @ U in registers ----
#pragma unroll
    for (int u = 0; u < 2; ++u) {
      const int tt = tt0 + u;
      const int tm = (tt >> 2) * 16, tn = (tt & 3) * 16;
      v8f co = {};
#pragma unroll
      for (int k0 = 0; k0 < Ddim; k0 += 32) {
        co = wmma_bf16(ldA(sPb, tm, k0, lane), ldB(sUt, k0, tn, lane), co);
        co = wmma_bf16(ldA(sQb, tm, k0, lane), ldB(sSt, k0, tn, lane), co);
      }
      const int n  = tn + (lane & 15);
      const int mb = tm + ((lane >> 4) << 3);
#pragma unroll
      for (int r = 0; r < 8; ++r)
        oh[(size_t)(t0 + mb + r) * Ddim + n] = co[r];

#pragma unroll
      for (int k0 = 0; k0 < Ddim; k0 += 32)
        sacc[u] = wmma_bf16(ldA(sKt, tm, k0, lane), ldB(sUt, k0, tn, lane), sacc[u]);
    }
    __syncthreads();  // all reads of sSt done before republishing S^T

    // publish bf16 S^T: lane writes 8 contiguous bf16 (one 16B store)
#pragma unroll
    for (int u = 0; u < 2; ++u) {
      const int tt = tt0 + u;
      const int tm = (tt >> 2) * 16, tn = (tt & 3) * 16;
      const int n  = tn + (lane & 15);
      const int mb = tm + ((lane >> 4) << 3);
      v8bf st;
#pragma unroll
      for (int r = 0; r < 8; ++r) st[r] = (__bf16)sacc[u][r];
      *reinterpret_cast<v8bf*>(&sSt[n * LDB + mb]) = st;
    }
    __syncthreads();
  }
}

extern "C" void kernel_launch(void* const* d_in, const int* in_sizes, int n_in,
                              void* d_out, int out_size, void* d_ws, size_t ws_size,
                              hipStream_t stream) {
  (void)in_sizes; (void)n_in; (void)out_size; (void)d_ws; (void)ws_size;
  const float* q    = (const float*)d_in[0];
  const float* k    = (const float*)d_in[1];
  const float* v    = (const float*)d_in[2];
  const float* beta = (const float*)d_in[3];
  float* out = (float*)d_out;
  delta_rule_chunked_kernel<<<dim3(Bdim * Hdim), dim3(256), 0, stream>>>(q, k, v, beta, out);
}
